// EnhancedGCN_51170240364592
// MI455X (gfx1250) — compile-verified, matrix operand order checked by reading, not run on hardware
//
#include <hip/hip_runtime.h>
#include <hip/hip_bf16.h>

// ---------------------------------------------------------------------------
// EnhancedGCN on MI455X (gfx1250).
//   GEMMs: bf16 operands -> v_wmma_f32_16x16x32_bf16, fragments loaded as
//          aligned global_load_b128 clauses (2 per fragment half).
//   Edge aggregation: precomputed per-edge norm + float4 gather +
//          native global_atomic_add_f32 (inline asm, no-return form).
//   Pointwise (BN+ReLU+bias+self-loop) fuses the bf16 down-convert for the
//   next layer's GEMM input (saves a full 51MB re-read pass per layer).
// Roofline: scatter dominates (~512MB + 64M atomics / layer); GEMM operands
// stay L2/WGP$-resident (x_bf16 = 25.6MB << 192MB L2).
// ---------------------------------------------------------------------------

#define NN    100000
#define EE    500000
#define F_IN  64
#define HH    128
#define EPSV  1e-5f

typedef __attribute__((ext_vector_type(16))) __bf16 v16bf;
typedef __attribute__((ext_vector_type(8)))  __bf16 v8bf;
typedef __attribute__((ext_vector_type(4)))  __bf16 v4bf;
typedef __attribute__((ext_vector_type(8)))  float  v8f;

// Native fp32 global atomic add, no-return form (tracked by STOREcnt).
__device__ __forceinline__ void atomic_add_f32(float* p, float v) {
    asm volatile("global_atomic_add_f32 %0, %1, off" : : "v"(p), "v"(v) : "memory");
}

// ---------------------------------------------------------------------------
// Utility kernels
// ---------------------------------------------------------------------------

__global__ void fill4_kernel(float4* __restrict__ p, float v, int n4) {
    int i = blockIdx.x * blockDim.x + threadIdx.x;
    if (i < n4) p[i] = make_float4(v, v, v, v);
}

// deg accumulation: deg[dst[e]] += 1 (deg pre-filled with 1.0 for self-loop)
__global__ void deg_kernel(const int* __restrict__ dst, float* __restrict__ deg) {
    int e = blockIdx.x * blockDim.x + threadIdx.x;
    if (e < EE) atomic_add_f32(&deg[dst[e]], 1.0f);
}

// in-place deg -> dinv = rsqrt(deg)
__global__ void dinv_kernel(float* __restrict__ d) {
    int i = blockIdx.x * blockDim.x + threadIdx.x;
    if (i < NN) d[i] = __frsqrt_rn(d[i]);
}

// per-edge symmetric norm
__global__ void norm_kernel(const int* __restrict__ src, const int* __restrict__ dst,
                            const float* __restrict__ dinv, float* __restrict__ norm) {
    int e = blockIdx.x * blockDim.x + threadIdx.x;
    if (e < EE) norm[e] = dinv[src[e]] * dinv[dst[e]];
}

// fp32 -> bf16 (same layout), 4 elements/thread (layer-1 input only)
__global__ void cvt_bf16_kernel(const float* __restrict__ in, __bf16* __restrict__ out, int n4) {
    int i = blockIdx.x * blockDim.x + threadIdx.x;
    if (i >= n4) return;
    const float4 f = ((const float4*)in)[i];
    v4bf r;
    r[0] = (__bf16)f.x; r[1] = (__bf16)f.y; r[2] = (__bf16)f.z; r[3] = (__bf16)f.w;
    ((v4bf*)out)[i] = r;
}

// W[K x H] fp32 -> WT[H x K] bf16 (column of W becomes contiguous row)
template <int K>
__global__ void cvt_wT_kernel(const float* __restrict__ W, __bf16* __restrict__ WT) {
    int i = blockIdx.x * blockDim.x + threadIdx.x;   // over H*K
    if (i >= HH * K) return;
    const int col = i / K;
    const int k   = i % K;
    WT[col * K + k] = (__bf16)W[k * HH + col];
}

// ---------------------------------------------------------------------------
// WMMA GEMM: Hout[N x 128] = Xb[N x K](bf16) * WTb[128 x K](bf16, transposed)
// Block = 256 threads = 8 waves; block b -> rows [16b,16b+16), wave w -> cols
// [16w,16w+16). N % 16 == 0 -> exact tiling, EXEC all-ones for every WMMA.
//
// bf16 fragment layout (CDNA5 ISA): lane (l<16) row/col = l, K chunk
// {kHi..kHi+7} in VGPR0-3 and {16+kHi..23+kHi} in VGPR4-7, kHi = (l>=16)*8.
// Each half is 8 contiguous bf16 = one aligned global_load_b128.
// ---------------------------------------------------------------------------
template <int K>
__global__ void gemm_wmma_kernel(const __bf16* __restrict__ Xb,
                                 const __bf16* __restrict__ WTb,
                                 float* __restrict__ Hout) {
    const int lane    = threadIdx.x & 31;
    const int wave    = threadIdx.x >> 5;
    const int rowBase = blockIdx.x << 4;
    const int m16     = lane & 15;
    const int row     = rowBase + m16;
    const int col     = (wave << 4) + m16;
    const int kHi     = (lane >> 4) << 3;      // 0 or 8

    const __bf16* xrow = Xb  + (size_t)row * K;
    const __bf16* wcol = WTb + (size_t)col * K;

    v8f acc = {};
#pragma unroll
    for (int kk = 0; kk < K; kk += 32) {
        const v8bf a0 = *(const v8bf*)(xrow + kk + kHi);
        const v8bf a1 = *(const v8bf*)(xrow + kk + 16 + kHi);
        const v8bf b0 = *(const v8bf*)(wcol + kk + kHi);
        const v8bf b1 = *(const v8bf*)(wcol + kk + 16 + kHi);
        v16bf a, b;
#pragma unroll
        for (int i = 0; i < 8; ++i) {
            a[i] = a0[i]; a[8 + i] = a1[i];
            b[i] = b0[i]; b[8 + i] = b1[i];
        }
        acc = __builtin_amdgcn_wmma_f32_16x16x32_bf16(
            false, a, false, b, (short)0, acc, false, false);
    }

    // C/D layout: lanes 0-15 -> M = r, lanes 16-31 -> M = 8 + r; N = lane&15.
    const int mTop = (lane < 16) ? 0 : 8;
#pragma unroll
    for (int r = 0; r < 8; ++r)
        Hout[(size_t)(rowBase + mTop + r) * HH + col] = acc[r];
}

// ---------------------------------------------------------------------------
// Edge scatter: agg[dst, :] += norm[e] * h[src, :]
// 32 threads/edge, 4 features each: one float4 gather + 4 native atomics.
// ---------------------------------------------------------------------------
__global__ void scatter_kernel(const float* __restrict__ h,
                               const int* __restrict__ src,
                               const int* __restrict__ dst,
                               const float* __restrict__ norm,
                               float* __restrict__ agg) {
    const int gid = blockIdx.x * blockDim.x + threadIdx.x;
    const int e   = gid >> 5;
    const int f4  = (gid & 31) << 2;
    if (e >= EE) return;
    const int   s = src[e];
    const int   d = dst[e];
    const float w = norm[e];
    const float4 hv = *(const float4*)(h + (size_t)s * HH + f4);
    float* base = agg + (size_t)d * HH + f4;
    atomic_add_f32(base + 0, w * hv.x);
    atomic_add_f32(base + 1, w * hv.y);
    atomic_add_f32(base + 2, w * hv.z);
    atomic_add_f32(base + 3, w * hv.w);
}

// ---------------------------------------------------------------------------
// Pointwise: out = relu(bn(agg + h*self_norm + bias)) [+ resid], float4-wide.
// WB: also emit bf16 copy (next layer's WMMA input) -- fused down-convert.
// ---------------------------------------------------------------------------
template <bool RES, bool WB>
__global__ void pointwise_kernel(const float* __restrict__ agg,
                                 const float* __restrict__ h,
                                 const float* __restrict__ dinv,
                                 const float* __restrict__ bias,
                                 const float* __restrict__ g,
                                 const float* __restrict__ be,
                                 const float* __restrict__ m,
                                 const float* __restrict__ v,
                                 const float* __restrict__ resid,
                                 float* __restrict__ out,
                                 __bf16* __restrict__ outb) {
    const int gid = blockIdx.x * blockDim.x + threadIdx.x;
    const int n   = gid >> 5;
    const int f   = (gid & 31) << 2;
    if (n >= NN) return;
    const size_t idx = (size_t)n * HH + f;
    const float sn = dinv[n] * dinv[n];
    const float4 A  = *(const float4*)(agg + idx);
    const float4 Hv = *(const float4*)(h + idx);
    const float4 B  = *(const float4*)(bias + f);
    const float4 G  = *(const float4*)(g + f);
    const float4 BE = *(const float4*)(be + f);
    const float4 M  = *(const float4*)(m + f);
    const float4 V  = *(const float4*)(v + f);
    float4 o;
    o.x = fmaxf((A.x + Hv.x * sn + B.x - M.x) * (G.x * __frsqrt_rn(V.x + EPSV)) + BE.x, 0.0f);
    o.y = fmaxf((A.y + Hv.y * sn + B.y - M.y) * (G.y * __frsqrt_rn(V.y + EPSV)) + BE.y, 0.0f);
    o.z = fmaxf((A.z + Hv.z * sn + B.z - M.z) * (G.z * __frsqrt_rn(V.z + EPSV)) + BE.z, 0.0f);
    o.w = fmaxf((A.w + Hv.w * sn + B.w - M.w) * (G.w * __frsqrt_rn(V.w + EPSV)) + BE.w, 0.0f);
    if (RES) {
        const float4 R = *(const float4*)(resid + idx);
        o.x += R.x; o.y += R.y; o.z += R.z; o.w += R.w;
    }
    *(float4*)(out + idx) = o;
    if (WB) {
        v4bf r;
        r[0] = (__bf16)o.x; r[1] = (__bf16)o.y; r[2] = (__bf16)o.z; r[3] = (__bf16)o.w;
        *(v4bf*)(outb + idx) = r;
    }
}

// ---------------------------------------------------------------------------
// Head: GEMV (H=128 -> 1), scalar scatter, final pointwise
// ---------------------------------------------------------------------------
__global__ void gemv_kernel(const float* __restrict__ X,
                            const float* __restrict__ W4,
                            float* __restrict__ h4) {
    const int n = blockIdx.x * blockDim.x + threadIdx.x;
    if (n >= NN) return;
    const float4* xr = (const float4*)(X + (size_t)n * HH);
    const float4* wr = (const float4*)W4;
    float s = 0.0f;
#pragma unroll
    for (int i = 0; i < HH / 4; ++i) {
        const float4 a = xr[i];
        const float4 b = wr[i];
        s += a.x * b.x + a.y * b.y + a.z * b.z + a.w * b.w;
    }
    h4[n] = s;
}

__global__ void scatter1_kernel(const float* __restrict__ h4,
                                const int* __restrict__ src,
                                const int* __restrict__ dst,
                                const float* __restrict__ norm,
                                float* __restrict__ agg4) {
    const int e = blockIdx.x * blockDim.x + threadIdx.x;
    if (e >= EE) return;
    atomic_add_f32(&agg4[dst[e]], norm[e] * h4[src[e]]);
}

__global__ void head_kernel(const float* __restrict__ agg4,
                            const float* __restrict__ h4,
                            const float* __restrict__ dinv,
                            const float* __restrict__ b4,
                            float* __restrict__ out) {
    const int n = blockIdx.x * blockDim.x + threadIdx.x;
    if (n >= NN) return;
    out[n] = agg4[n] + h4[n] * dinv[n] * dinv[n] + b4[0];
}

// ---------------------------------------------------------------------------
// Launch
// ---------------------------------------------------------------------------
extern "C" void kernel_launch(void* const* d_in, const int* in_sizes, int n_in,
                              void* d_out, int out_size, void* d_ws, size_t ws_size,
                              hipStream_t stream) {
    const float* x   = (const float*)d_in[0];
    const int*   ei  = (const int*)d_in[1];
    const float* W1  = (const float*)d_in[2];
    const float* b1  = (const float*)d_in[3];
    const float* g1  = (const float*)d_in[4];
    const float* be1 = (const float*)d_in[5];
    const float* m1  = (const float*)d_in[6];
    const float* v1  = (const float*)d_in[7];
    const float* W2  = (const float*)d_in[8];
    const float* b2  = (const float*)d_in[9];
    const float* g2  = (const float*)d_in[10];
    const float* be2 = (const float*)d_in[11];
    const float* m2  = (const float*)d_in[12];
    const float* v2  = (const float*)d_in[13];
    const float* W3  = (const float*)d_in[14];
    const float* b3  = (const float*)d_in[15];
    const float* g3  = (const float*)d_in[16];
    const float* be3 = (const float*)d_in[17];
    const float* m3  = (const float*)d_in[18];
    const float* v3  = (const float*)d_in[19];
    const float* W4  = (const float*)d_in[20];
    const float* b4  = (const float*)d_in[21];

    const int* src = ei;           // edge_index[0]
    const int* dst = ei + EE;      // edge_index[1]

    // Workspace layout:
    //  floats: dinv[N] | norm[E] | h[N*H] | agg[N*H] | x1[N*H] | xcur[N*H]
    //  bf16  : xb[N*H] | wtb[H*H]
    float* ws    = (float*)d_ws;
    float* dinv  = ws;
    float* norm  = dinv + NN;
    float* h     = norm + EE;
    float* agg   = h    + (size_t)NN * HH;
    float* x1    = agg  + (size_t)NN * HH;
    float* xcur  = x1   + (size_t)NN * HH;
    __bf16* xb   = (__bf16*)(xcur + (size_t)NN * HH);
    __bf16* wtb  = xb + (size_t)NN * HH;
    float* h4    = h;     // head reuses h
    float* agg4  = agg;   // head reuses agg

    const int T = 256;
    const dim3 blk(T);
    const dim3 gN((NN + T - 1) / T);                 // per-node
    const dim3 gE((EE + T - 1) / T);                 // per-edge
    const dim3 gNH4((NN * HH / 4 + T - 1) / T);      // per-(node,4feat): 12500
    const dim3 gEH4((EE * 32 + T - 1) / T);          // scatter: 62500
    const dim3 gTile(NN / 16);                       // GEMM strips: 6250
    const dim3 gW64((HH * F_IN + T - 1) / T);
    const dim3 gW128((HH * HH + T - 1) / T);
    const dim3 gCvt64((NN * F_IN / 4 + T - 1) / T);

    // degree / norms
    fill4_kernel<<<dim3(NN / 4 / T + 1), blk, 0, stream>>>((float4*)dinv, 1.0f, NN / 4);
    deg_kernel<<<gE, blk, 0, stream>>>(dst, dinv);
    dinv_kernel<<<gN, blk, 0, stream>>>(dinv);
    norm_kernel<<<gE, blk, 0, stream>>>(src, dst, dinv, norm);

    // ---- layer 1 (K = 64) ----
    cvt_bf16_kernel<<<gCvt64, blk, 0, stream>>>(x, xb, NN * F_IN / 4);
    cvt_wT_kernel<F_IN><<<gW64, blk, 0, stream>>>(W1, wtb);
    fill4_kernel<<<gNH4, blk, 0, stream>>>((float4*)agg, 0.0f, NN * HH / 4);
    gemm_wmma_kernel<F_IN><<<gTile, blk, 0, stream>>>(xb, wtb, h);
    scatter_kernel<<<gEH4, blk, 0, stream>>>(h, src, dst, norm, agg);
    // x1 (fp32, kept for residual) + xb (bf16, layer-2 GEMM input), fused
    pointwise_kernel<false, true><<<gNH4, blk, 0, stream>>>(agg, h, dinv, b1, g1, be1, m1, v1, nullptr, x1, xb);

    // ---- layer 2 (K = 128) ----
    cvt_wT_kernel<HH><<<gW128, blk, 0, stream>>>(W2, wtb);
    fill4_kernel<<<gNH4, blk, 0, stream>>>((float4*)agg, 0.0f, NN * HH / 4);
    gemm_wmma_kernel<HH><<<gTile, blk, 0, stream>>>(xb, wtb, h);
    scatter_kernel<<<gEH4, blk, 0, stream>>>(h, src, dst, norm, agg);
    // x2 fp32 (not reused later as fp32) + bf16 for layer-3 GEMM
    pointwise_kernel<false, true><<<gNH4, blk, 0, stream>>>(agg, h, dinv, b2, g2, be2, m2, v2, nullptr, xcur, xb);

    // ---- layer 3 (K = 128, + residual x1) ----
    cvt_wT_kernel<HH><<<gW128, blk, 0, stream>>>(W3, wtb);
    fill4_kernel<<<gNH4, blk, 0, stream>>>((float4*)agg, 0.0f, NN * HH / 4);
    gemm_wmma_kernel<HH><<<gTile, blk, 0, stream>>>(xb, wtb, h);
    scatter_kernel<<<gEH4, blk, 0, stream>>>(h, src, dst, norm, agg);
    // x3 fp32 only (feeds fp32 GEMV head; no bf16 copy needed)
    pointwise_kernel<true, false><<<gNH4, blk, 0, stream>>>(agg, h, dinv, b3, g3, be3, m3, v3, x1, xcur, nullptr);

    // ---- head (H -> 1) ----
    gemv_kernel<<<gN, blk, 0, stream>>>(xcur, W4, h4);
    fill4_kernel<<<dim3(NN / 4 / T + 1), blk, 0, stream>>>((float4*)agg4, 0.0f, NN / 4);
    scatter1_kernel<<<gE, blk, 0, stream>>>(h4, src, dst, norm, agg4);
    head_kernel<<<gN, blk, 0, stream>>>(agg4, h4, dinv, b4, (float*)d_out);
}
